// GAT_79328045957507
// MI455X (gfx1250) — matrix operand order repaired and checked by reading, not compiled, not run on hardware
//
#include <hip/hip_runtime.h>
#include <hip/hip_bf16.h>

// ---------------------------------------------------------------------------
// CDNA5 (gfx1250, wave32) fused GAT forward.
//   - v_wmma_f32_16x16x32_bf16 for all matmuls (feature GEMMs + att@Wh)
//   - flash-style two-pass softmax; probabilities generated directly in the
//     WMMA A-fragment layout (lane l holds row l%16, K per ISA 16-bit A map)
//   - operands pre-transposed to bf16 so B fragments are contiguous b128 loads
//   - LDS staging via GLOBAL_LOAD_ASYNC_TO_LDS_B128 (+ s_wait_asynccnt)
//   - packed f32->bf16 via v_cvt_pk_bf16_f32
//   - 4-wide vectorized adj/Wh2 streaming in the softmax loops
// ---------------------------------------------------------------------------

typedef __attribute__((ext_vector_type(4)))  int            v4i;
typedef __attribute__((ext_vector_type(4)))  float          v4f;
typedef __attribute__((ext_vector_type(8)))  float          v8f;
typedef __attribute__((ext_vector_type(2)))  __bf16         v2bf;
typedef __attribute__((ext_vector_type(16))) __bf16         v16bf;
typedef __attribute__((ext_vector_type(8)))  unsigned short v8us;
typedef __attribute__((ext_vector_type(16))) unsigned short v16us;

#define NEG_BIG (-9.0e15f)
#define SLOPE   0.2f

__device__ __forceinline__ v2bf pk_bf16(float a, float b) {
#if __has_builtin(__builtin_amdgcn_cvt_pk_bf16_f32)
    return __builtin_amdgcn_cvt_pk_bf16_f32(a, b);
#else
    unsigned r;
    asm("v_cvt_pk_bf16_f32 %0, %1, %2" : "=v"(r) : "v"(a), "v"(b));
    return __builtin_bit_cast(v2bf, r);
#endif
}

__device__ __forceinline__ v16bf cat16(v8us lo, v8us hi) {
    v16us u = __builtin_shufflevector(lo, hi, 0, 1, 2, 3, 4, 5, 6, 7,
                                      8, 9, 10, 11, 12, 13, 14, 15);
    return __builtin_bit_cast(v16bf, u);
}

__device__ __forceinline__ float lrelu(float v) { return v > 0.f ? v : SLOPE * v; }
__device__ __forceinline__ float elu1(float v)  { return v > 0.f ? v : expm1f(v); }

// LDS byte offset of a generic pointer to __shared__ (hw uses addr[31:0])
__device__ __forceinline__ unsigned lds_off(const void* p) {
    return (unsigned)(unsigned long long)p;
}

// gfx1250 async global->LDS copy, 16B per active lane; tracked by ASYNCcnt
__device__ __forceinline__ void async_b128(unsigned ldsoff, const void* gaddr) {
    asm volatile("global_load_async_to_lds_b128 %0, %1, off"
                 :: "v"(ldsoff), "v"(gaddr) : "memory");
}
__device__ __forceinline__ void async_wait0() {
    asm volatile("s_wait_asynccnt 0" ::: "memory");
}

// ---------------------------------------------------------------------------
// graph_norm (faithful raw-reshape semantics): one block per (b, channel),
// normalizing a contiguous 2048-float chunk of the flat tensor.
// ---------------------------------------------------------------------------
__global__ void gn_kernel(const float* __restrict__ x,
                          const float* __restrict__ alpha,
                          const float* __restrict__ scale,
                          const float* __restrict__ shift,
                          float* __restrict__ out) {
    int b = blockIdx.x >> 7;
    int c = blockIdx.x & 127;
    const float* src = x   + ((size_t)b * 128 + c) * 2048;
    float*       dst = out + ((size_t)b * 128 + c) * 2048;
    int t = threadIdx.x;

    float v[8];
    float s = 0.f, s2 = 0.f;
#pragma unroll
    for (int i = 0; i < 8; ++i) {
        float xv = src[t + 256 * i];
        v[i] = xv; s += xv; s2 += xv * xv;
    }
    __shared__ float red[256], red2[256];
    red[t] = s; red2[t] = s2;
    __syncthreads();
    for (int off = 128; off > 0; off >>= 1) {
        if (t < off) { red[t] += red[t + off]; red2[t] += red2[t + off]; }
        __syncthreads();
    }
    float mean = red[0]  * (1.f / 2048.f);
    float var  = red2[0] * (1.f / 2048.f) - mean * mean;
    float inv  = rsqrtf(var + 1e-5f);
    float am = alpha[c] * mean, sc = scale[c], sh = shift[c];
#pragma unroll
    for (int i = 0; i < 8; ++i)
        dst[t + 256 * i] = (v[i] - am) * inv * sc + sh;
}

// ---------------------------------------------------------------------------
// Transpose + f32->bf16 convert: in [S][R][C] f32 -> out [S][C][R] bf16.
// ---------------------------------------------------------------------------
__global__ void tcvt_kernel(const float* __restrict__ in,
                            unsigned short* __restrict__ out,
                            int rsh /*log2(R/2)*/, int cmask /*C-1*/, int csh /*log2 C*/) {
    int idx = blockIdx.x * blockDim.x + threadIdx.x;
    int sc = idx >> rsh;
    int r2 = (idx & ((1 << rsh) - 1)) * 2;
    int s  = sc >> csh;
    int c  = sc & cmask;
    int C  = cmask + 1;
    int R  = 1 << (rsh + 1);
    const float* src = in + ((size_t)s * R + r2) * C + c;
    v2bf p = pk_bf16(src[0], src[C]);
    *(unsigned*)&out[((size_t)s * C + c) * R + r2] = __builtin_bit_cast(unsigned, p);
}

// ---------------------------------------------------------------------------
// C[M,128] = A[M,128] @ W[128,128]; WT is W pre-transposed bf16 [n][k].
// ---------------------------------------------------------------------------
__global__ void gemm128_kernel(const float* __restrict__ A,
                               const unsigned short* __restrict__ WT,
                               float* __restrict__ C) {
    extern __shared__ char dynsmem[];
    unsigned short* as = (unsigned short*)dynsmem;   // 16x128 bf16 A tile
    int t = threadIdx.x;
    int rowBase = blockIdx.x * 16;
    const float* Ab = A + (size_t)rowBase * 128;
    for (int i = t; i < 1024; i += 256)
        ((unsigned*)as)[i] = __builtin_bit_cast(unsigned, pk_bf16(Ab[2 * i], Ab[2 * i + 1]));
    __syncthreads();

    int wave = t >> 5, lane = t & 31;
    int half = lane >> 4, lr = lane & 15;
    const unsigned short* arow = as + lr * 128;
    const unsigned short* wcol = WT + (size_t)(wave * 16 + lr) * 128;
    v8f acc = {};
#pragma unroll
    for (int kt = 0; kt < 4; ++kt) {
        v8us a0 = *(const v8us*)(arow + kt * 32 + half * 8);
        v8us a1 = *(const v8us*)(arow + kt * 32 + 16 + half * 8);
        v8us b0 = *(const v8us*)(wcol + kt * 32 + half * 8);
        v8us b1 = *(const v8us*)(wcol + kt * 32 + 16 + half * 8);
        acc = __builtin_amdgcn_wmma_f32_16x16x32_bf16(
                false, cat16(a0, a1), false, cat16(b0, b1),
                (short)0, acc, false, false);
    }
#pragma unroll
    for (int r = 0; r < 8; ++r) {
        int row = (half ? 8 : 0) + r;
        C[(size_t)(rowBase + row) * 128 + wave * 16 + lr] = acc[r];
    }
}

// ---------------------------------------------------------------------------
// Attention projections: Wh1 = Wh·a[:k], Wh2 = Wh·a[k:]
// ---------------------------------------------------------------------------
__global__ void proj16_kernel(const float* __restrict__ Wh, const float* __restrict__ a,
                              float* __restrict__ w1, float* __restrict__ w2) {
    int i = blockIdx.x * blockDim.x + threadIdx.x;   // (b*8+h)*2048+n
    const float* p = Wh + (size_t)i * 16;
    float s1 = 0.f, s2 = 0.f;
#pragma unroll
    for (int j = 0; j < 16; ++j) { float v = p[j]; s1 += v * a[j]; s2 += v * a[16 + j]; }
    w1[i] = s1; w2[i] = s2;
}

__global__ void proj128_kernel(const float* __restrict__ Wh, const float* __restrict__ a,
                               float* __restrict__ w1, float* __restrict__ w2) {
    int i = blockIdx.x * blockDim.x + threadIdx.x;   // b*2048+n
    const float* p = Wh + (size_t)i * 128;
    float s1 = 0.f, s2 = 0.f;
    for (int j = 0; j < 128; ++j) { float v = p[j]; s1 += v * a[j]; s2 += v * a[128 + j]; }
    w1[i] = s1; w2[i] = s2;
}

// ---------------------------------------------------------------------------
// Online-softmax pass 1 helper: 4-wide chunk update.
// ---------------------------------------------------------------------------
__device__ __forceinline__ void pass1_chunk(const int* __restrict__ adjp,
                                            const float* __restrict__ whp,
                                            float wh1v, float& m, float& s) {
    v4i a4 = *(const v4i*)adjp;
    v4f w4 = *(const v4f*)whp;
    float v0 = a4[0] > 0 ? lrelu(wh1v + w4[0]) : NEG_BIG;
    float v1 = a4[1] > 0 ? lrelu(wh1v + w4[1]) : NEG_BIG;
    float v2 = a4[2] > 0 ? lrelu(wh1v + w4[2]) : NEG_BIG;
    float v3 = a4[3] > 0 ? lrelu(wh1v + w4[3]) : NEG_BIG;
    float m4 = fmaxf(fmaxf(v0, v1), fmaxf(v2, v3));
    float nm = fmaxf(m, m4);
    s = s * __expf(m - nm)
      + __expf(v0 - nm) + __expf(v1 - nm) + __expf(v2 - nm) + __expf(v3 - nm);
    m = nm;
}

// Pass-2 helper: build the 16-value probability A-fragment for one 32-wide
// j-tile.  Per lane the needed j's are two contiguous 8-runs starting at
// jt*32 + half*8 and jt*32 + half*8 + 16 (ISA 16-bit A-fragment layout).
__device__ __forceinline__ v16bf make_afrag(const int* __restrict__ ap,
                                            const float* __restrict__ wp,
                                            float wh1v, float M, float invS) {
    v4i aA = *(const v4i*)ap,        aB = *(const v4i*)(ap + 4);
    v4i aC = *(const v4i*)(ap + 16), aD = *(const v4i*)(ap + 20);
    v4f wA = *(const v4f*)wp,        wB = *(const v4f*)(wp + 4);
    v4f wC = *(const v4f*)(wp + 16), wD = *(const v4f*)(wp + 20);
    float wv[16]; int av[16];
#pragma unroll
    for (int q = 0; q < 4; ++q) {
        wv[q] = wA[q]; wv[4 + q] = wB[q]; wv[8 + q] = wC[q]; wv[12 + q] = wD[q];
        av[q] = aA[q]; av[4 + q] = aB[q]; av[8 + q] = aC[q]; av[12 + q] = aD[q];
    }
    v16bf au;
#pragma unroll
    for (int v = 0; v < 8; ++v) {
        int k0 = 2 * v;
        float e0 = lrelu(wh1v + wv[k0]);
        float e1 = lrelu(wh1v + wv[k0 + 1]);
        float v0 = av[k0]     > 0 ? e0 : NEG_BIG;
        float v1 = av[k0 + 1] > 0 ? e1 : NEG_BIG;
        v2bf pk = pk_bf16(__expf(v0 - M) * invS, __expf(v1 - M) * invS);
        au[2 * v] = pk[0]; au[2 * v + 1] = pk[1];
    }
    return au;
}

// ---------------------------------------------------------------------------
// Layer-1 attention (H=8, nh=16), flash-style, concat + ELU fused.
// grid = B*H*16 row-blocks; block = 256 thr (8 waves x 16 rows).
// LDS: Wh2 vector (8KB f32) + WhT tile [16][2048] bf16 (64KB), async-staged.
// ---------------------------------------------------------------------------
__global__ void attn1_kernel(const unsigned short* __restrict__ WhT, // [B*H][16][2048] bf16
                             const float* __restrict__ w1,
                             const float* __restrict__ w2,
                             const int*   __restrict__ adj,
                             float* __restrict__ outp) {
    extern __shared__ char dynsmem[];
    float*          wh2s = (float*)dynsmem;                        // 2048 f32
    unsigned short* whsT = (unsigned short*)(dynsmem + 8192);      // [16][2048] bf16

    int bid = blockIdx.x;
    int b = bid >> 7;
    int h = (bid >> 4) & 7;
    int rowBlk = bid & 15;
    int t = threadIdx.x;

    const char* gsrc  = (const char*)(WhT + ((size_t)(b * 8 + h)) * 32768);
    const char* g2src = (const char*)(w2 + ((size_t)b * 8 + h) * 2048);
    unsigned vbase = lds_off(wh2s), wbase = lds_off(whsT);
#pragma unroll
    for (int k = 0; k < 2; ++k) {                 // 8KB Wh2 vector
        int off = (k * 256 + t) * 16;
        async_b128(vbase + off, g2src + off);
    }
#pragma unroll
    for (int k = 0; k < 16; ++k) {                // 64KB WhT tile
        int off = (k * 256 + t) * 16;
        async_b128(wbase + off, gsrc + off);
    }
    async_wait0();
    __syncthreads();

    int wave = t >> 5, lane = t & 31;
    int half = lane >> 4, lr = lane & 15;
    int row  = rowBlk * 128 + wave * 16 + lr;
    float wh1v = w1[((size_t)b * 8 + h) * 2048 + row];
    const int* adjrow = adj + ((size_t)b * 2048 + row) * 2048;

    // pass 1: online max / exp-sum, 4-wide (halves interleave 4-chunks)
    float m = -3.4e38f, s = 0.f;
    for (int it = 0; it < 256; ++it) {
        int j = (it * 2 + half) * 4;
        pass1_chunk(adjrow + j, wh2s + j, wh1v, m, s);
    }
    float mo = __shfl_xor(m, 16, 32);
    float so = __shfl_xor(s, 16, 32);
    float M = fmaxf(m, mo);
    float S = s * __expf(m - M) + so * __expf(mo - M);
    float invS = 1.f / S;

    // pass 2: probabilities in A-fragment layout, WMMA accumulate
    v8f acc = {};
    for (int jt = 0; jt < 64; ++jt) {
        v16bf au = make_afrag(adjrow + jt * 32 + half * 8,
                              wh2s   + jt * 32 + half * 8, wh1v, M, invS);
        const unsigned short* bp = whsT + lr * 2048 + jt * 32 + half * 8;
        v8us b0 = *(const v8us*)bp;
        v8us b1 = *(const v8us*)(bp + 16);
        acc = __builtin_amdgcn_wmma_f32_16x16x32_bf16(
                false, au, false, cat16(b0, b1), (short)0, acc, false, false);
    }

    int nbase = rowBlk * 128 + wave * 16;
#pragma unroll
    for (int r = 0; r < 8; ++r) {
        int rown = nbase + (half ? 8 : 0) + r;
        outp[(size_t)b * 262144 + (size_t)h * 32768 + (size_t)rown * 16 + lr] = elu1(acc[r]);
    }
}

// ---------------------------------------------------------------------------
// Layer-2 attention (H=1, nh=128), mean-over-1-head + ELU fused -> d_out.
// grid = B*16; block = 256 thr (8 waves x 16 rows, each wave 16x128 output).
// LDS: Wh2 vector (8KB) + streamed transposed tile [128][32] bf16 (8KB), async.
// ---------------------------------------------------------------------------
__global__ void attn2_kernel(const unsigned short* __restrict__ WhT, // [B][128][2048] bf16
                             const float* __restrict__ w1,
                             const float* __restrict__ w2,
                             const int*   __restrict__ adj,
                             float* __restrict__ outp) {
    extern __shared__ char dynsmem[];
    float*          wh2s  = (float*)dynsmem;                   // 2048 f32
    unsigned short* tileT = (unsigned short*)(dynsmem + 8192); // [128][32] bf16

    int b = blockIdx.x >> 4;
    int rowBlk = blockIdx.x & 15;
    int t = threadIdx.x;

    unsigned vbase = lds_off(wh2s), tbase = lds_off(tileT);
    const char* g2src = (const char*)(w2 + (size_t)b * 2048);
#pragma unroll
    for (int k = 0; k < 2; ++k) {
        int off = (k * 256 + t) * 16;
        async_b128(vbase + off, g2src + off);
    }
    async_wait0();
    __syncthreads();

    int wave = t >> 5, lane = t & 31;
    int half = lane >> 4, lr = lane & 15;
    int row  = rowBlk * 128 + wave * 16 + lr;
    float wh1v = w1[(size_t)b * 2048 + row];
    const int* adjrow = adj + ((size_t)b * 2048 + row) * 2048;

    float m = -3.4e38f, s = 0.f;
    for (int it = 0; it < 256; ++it) {
        int j = (it * 2 + half) * 4;
        pass1_chunk(adjrow + j, wh2s + j, wh1v, m, s);
    }
    float mo = __shfl_xor(m, 16, 32);
    float so = __shfl_xor(s, 16, 32);
    float M = fmaxf(m, mo);
    float S = s * __expf(m - M) + so * __expf(mo - M);
    float invS = 1.f / S;

    const char* gT = (const char*)(WhT + (size_t)b * 262144);
    v8f acc[8] = {};
    for (int jt = 0; jt < 64; ++jt) {
        __syncthreads();
#pragma unroll
        for (int k = 0; k < 2; ++k) {             // stage [128][32] bf16 tile
            int o = t * 2 + k;                    // 512 x 16B chunks
            int c = o >> 2, part = o & 3;
            async_b128(tbase + c * 64 + part * 16,
                       gT + ((size_t)c * 2048 + jt * 32) * 2 + part * 16);
        }
        async_wait0();
        __syncthreads();

        v16bf au = make_afrag(adjrow + jt * 32 + half * 8,
                              wh2s   + jt * 32 + half * 8, wh1v, M, invS);
#pragma unroll
        for (int cb = 0; cb < 8; ++cb) {
            const unsigned short* bp = tileT + (cb * 16 + lr) * 32 + half * 8;
            v8us b0 = *(const v8us*)bp;
            v8us b1 = *(const v8us*)(bp + 16);
            acc[cb] = __builtin_amdgcn_wmma_f32_16x16x32_bf16(
                        false, au, false, cat16(b0, b1), (short)0, acc[cb], false, false);
        }
    }

    int nbase = rowBlk * 128 + wave * 16;
#pragma unroll
    for (int cb = 0; cb < 8; ++cb) {
#pragma unroll
        for (int r = 0; r < 8; ++r) {
            int rown = nbase + (half ? 8 : 0) + r;
            outp[(size_t)b * 262144 + (size_t)rown * 128 + cb * 16 + lr] = elu1(acc[cb][r]);
        }
    }
}

// ---------------------------------------------------------------------------
// Host launcher
// ---------------------------------------------------------------------------
extern "C" void kernel_launch(void* const* d_in, const int* in_sizes, int n_in,
                              void* d_out, int out_size, void* d_ws, size_t ws_size,
                              hipStream_t stream) {
    (void)in_sizes; (void)n_in; (void)out_size; (void)ws_size;
    const float* x   = (const float*)d_in[0];
    const int*   adj = (const int*)  d_in[1];
    const float* W1  = (const float*)d_in[2];
    const float* a1  = (const float*)d_in[3];
    const float* g1a = (const float*)d_in[4];
    const float* g1s = (const float*)d_in[5];
    const float* g1b = (const float*)d_in[6];
    const float* W2  = (const float*)d_in[7];
    const float* a2  = (const float*)d_in[8];
    const float* g2a = (const float*)d_in[9];
    const float* g2s = (const float*)d_in[10];
    const float* g2b = (const float*)d_in[11];
    float* out = (float*)d_out;

    // workspace layout
    float* ws   = (float*)d_ws;
    float* h1   = ws;                 // 1048576  gn1 output
    float* Whb  = h1   + 1048576;     // 1048576  layer-1 Wh (f32)
    float* w1v  = Whb  + 1048576;     // 65536    layer-1 Wh1
    float* w2v  = w1v  + 65536;       // 65536    layer-1 Wh2
    float* hp1  = w2v  + 65536;       // 1048576  layer-1 output (post-ELU)
    float* h2   = hp1  + 1048576;     // 1048576  gn2 output
    float* Whb2 = h2   + 1048576;     // 1048576  layer-2 Wh (f32)
    float* w1v2 = Whb2 + 1048576;     // 8192     layer-2 Wh1
    float* w2v2 = w1v2 + 8192;        // 8192     layer-2 Wh2
    unsigned short* WT1b  = (unsigned short*)(w2v2 + 8192); // 16384  W1^T bf16
    unsigned short* WT2b  = WT1b  + 16384;                  // 16384  W2^T bf16
    unsigned short* WhT1b = WT2b  + 16384;                  // 1048576 Wh^T bf16 (L1)
    unsigned short* WhT2b = WhT1b + 1048576;                // 1048576 Wh^T bf16 (L2)

    dim3 blk(256);
    // pre-transpose weights to bf16 [n][k]
    tcvt_kernel   <<<32,   blk, 0, stream>>>(W1, WT1b, 6, 127, 7);
    tcvt_kernel   <<<32,   blk, 0, stream>>>(W2, WT2b, 6, 127, 7);

    gn_kernel     <<<512,  blk, 0,            stream>>>(x, g1a, g1s, g1b, h1);
    gemm128_kernel<<<512,  blk, 16 * 128 * 2, stream>>>(h1, WT1b, Whb);
    proj16_kernel <<<256,  blk, 0,            stream>>>(Whb, a1, w1v, w2v);
    tcvt_kernel   <<<2048, blk, 0,            stream>>>(Whb, WhT1b, 10, 15, 4);
    attn1_kernel  <<<512,  blk, 8192 + 65536, stream>>>(WhT1b, w1v, w2v, adj, hp1);

    gn_kernel     <<<512,  blk, 0,            stream>>>(hp1, g2a, g2s, g2b, h2);
    gemm128_kernel<<<512,  blk, 16 * 128 * 2, stream>>>(h2, WT2b, Whb2);
    proj128_kernel<<<32,   blk, 0,            stream>>>(Whb2, a2, w1v2, w2v2);
    tcvt_kernel   <<<2048, blk, 0,            stream>>>(Whb2, WhT2b, 10, 127, 7);
    attn2_kernel  <<<64,   blk, 8192 + 8192,  stream>>>(WhT2b, w1v2, w2v2, adj, out);
}
